// ALBEF_32959579029880
// MI455X (gfx1250) — compile-verified, matrix operand order checked by reading, not run on hardware
//
#include <hip/hip_runtime.h>
#include <hip/hip_bf16.h>
#include <math.h>

typedef _Float16 v16h __attribute__((ext_vector_type(16)));
typedef _Float16 v8h  __attribute__((ext_vector_type(8)));
typedef _Float16 v4h  __attribute__((ext_vector_type(4)));
typedef float    v8f  __attribute__((ext_vector_type(8)));
typedef float    v4f  __attribute__((ext_vector_type(4)));

#define B_    32
#define T_    60
#define I_    577
#define H_    768
#define P_    384
#define G_    256
#define C_    128
#define N_NODE 637      // I + T
#define HPAD   640      // padded node count (multiple of 32) for WMMA K loop
#define NEG_SLOPE 0.2f

// LDS tile row stride in halves: 40 -> 80B rows, 16B-aligned v8h loads, good bank spread
#define LSTR 40
#define ASTR 648        // attention-weight LDS row stride (halves), 1296B = 16B aligned

static __device__ __forceinline__ v16h make_v16(v8h lo, v8h hi) {
  return __builtin_shufflevector(lo, hi, 0,1,2,3,4,5,6,7,8,9,10,11,12,13,14,15);
}
static __device__ __forceinline__ v4h pack4(v4f v) {
  v4h r;
  r[0] = (_Float16)v[0]; r[1] = (_Float16)v[1];
  r[2] = (_Float16)v[2]; r[3] = (_Float16)v[3];
  return r;
}

// ---------------------------------------------------------------------------
// Generic f32-in / f32-out GEMM via f16 WMMA: out = act(A[M,K] @ W[K,N] + bias)
// Output rows remapped: b = r / Rin, rr = r % Rin,
//   out[(b*outStride + rowOff + rr)*ldOut + n]
// act: 0=none, 1=sigmoid, 2=relu
// Block: 256 threads (8 waves, 4x2), tile 128(M) x 128(N), K-step 32.
// Double-buffered LDS (1 barrier/tile); global prefetch before the barrier;
// next tile's LDS stores issued after the WMMAs so they overlap matrix math.
// W tile is loaded as coalesced k-strips so the transposed store is a b64.
// ---------------------------------------------------------------------------
__global__ __launch_bounds__(256) void gemm_act_kernel(
    const float* __restrict__ A, const float* __restrict__ W,
    const float* __restrict__ bias, float* __restrict__ out,
    int M, int N, int K, int act, int Rin, int outStride, int rowOff, int ldOut)
{
  __shared__ _Float16 As[2][128 * LSTR];   // A tile  [128 rows][32 k]
  __shared__ _Float16 Bs[2][128 * LSTR];   // W tile transposed [128 n][32 k]

  const int tid  = threadIdx.x;
  const int m0   = blockIdx.x * 128;
  const int n0   = blockIdx.y * 128;
  const int lane = tid & 31;
  const int wid  = tid >> 5;
  const int wm   = wid >> 1;            // 0..3 -> M offset wm*32
  const int wn   = wid & 1;             // 0..1 -> N offset wn*64

  const v8f vzero = {0.f,0.f,0.f,0.f,0.f,0.f,0.f,0.f};
  v8f acc[2][4];
  #pragma unroll
  for (int i = 0; i < 2; i++)
    #pragma unroll
    for (int j = 0; j < 4; j++) acc[i][j] = vzero;

  const int koff = (lane & 16) ? 8 : 0;
  const int l15  = lane & 15;

  // staging registers
  v4f aR[4];              // A: 4 float4 (row-quads)
  float bRk[4][4];        // W: 4 k-strips of 4 (coalesced along n per k row)
  const v4f z4 = {0.f, 0.f, 0.f, 0.f};

  auto load_tiles = [&](int k0) {
    #pragma unroll
    for (int it = 0; it < 4; it++) {          // A: 1024 float4, i -> (row, kquad)
      int i  = tid + it * 256;
      int r  = i >> 3, kq = i & 7;
      int gr = m0 + r;
      int rc = (gr < M) ? gr : (M - 1);       // clamp: unconditional load
      v4f v  = *(const v4f*)&A[(size_t)rc * K + k0 + kq * 4];
      aR[it] = (gr < M) ? v : z4;             // cndmask, no exec juggling
    }
    #pragma unroll
    for (int it = 0; it < 4; it++) {          // W: i -> (n, kquad); n fast => coalesced
      int i  = tid + it * 256;
      int n  = i & 127, kq = i >> 7;
      #pragma unroll
      for (int j = 0; j < 4; j++)
        bRk[it][j] = W[(size_t)(k0 + kq * 4 + j) * N + n0 + n];
    }
  };
  auto store_tiles = [&](int buf) {
    #pragma unroll
    for (int it = 0; it < 4; it++) {
      int i = tid + it * 256;
      int r = i >> 3, kq = i & 7;
      *(v4h*)&As[buf][r * LSTR + kq * 4] = pack4(aR[it]);        // b64 store
    }
    #pragma unroll
    for (int it = 0; it < 4; it++) {
      int i  = tid + it * 256;
      int n  = i & 127, kq = i >> 7;
      v4f t  = { bRk[it][0], bRk[it][1], bRk[it][2], bRk[it][3] };
      *(v4h*)&Bs[buf][n * LSTR + kq * 4] = pack4(t);             // b64 store
    }
  };

  const int KT = K >> 5;
  load_tiles(0);
  store_tiles(0);
  for (int t = 0; t < KT; t++) {
    const int cur = t & 1;
    if (t + 1 < KT) load_tiles((t + 1) << 5);   // global prefetch
    __syncthreads();                            // publish stores into buf[cur]

    const _Float16* Ab = As[cur];
    const _Float16* Bb = Bs[cur];
    v16h af[2], bf[4];
    #pragma unroll
    for (int ms = 0; ms < 2; ms++) {
      int row = wm * 32 + ms * 16 + l15;
      v8h lo = *(const v8h*)&Ab[row * LSTR + koff];
      v8h hi = *(const v8h*)&Ab[row * LSTR + 16 + koff];
      af[ms] = make_v16(lo, hi);
    }
    #pragma unroll
    for (int ns = 0; ns < 4; ns++) {
      int col = wn * 64 + ns * 16 + l15;
      v8h lo = *(const v8h*)&Bb[col * LSTR + koff];
      v8h hi = *(const v8h*)&Bb[col * LSTR + 16 + koff];
      bf[ns] = make_v16(lo, hi);
    }
    #pragma unroll
    for (int ms = 0; ms < 2; ms++)
      #pragma unroll
      for (int ns = 0; ns < 4; ns++)
        acc[ms][ns] = __builtin_amdgcn_wmma_f32_16x16x32_f16(
            false, af[ms], false, bf[ns], (short)0, acc[ms][ns], false, false);

    if (t + 1 < KT) store_tiles(1 - cur);       // overlaps WMMAs above
  }

  // epilogue: lane<16 -> col=lane, rows v; lane>=16 -> col=lane-16, rows 8+v
  const int mb = (lane & 16) ? 8 : 0;
  #pragma unroll
  for (int ms = 0; ms < 2; ms++) {
    #pragma unroll
    for (int ns = 0; ns < 4; ns++) {
      int n  = n0 + wn * 64 + ns * 16 + l15;
      float bvl = bias ? bias[n] : 0.f;
      int mbase = m0 + wm * 32 + ms * 16 + mb;
      #pragma unroll
      for (int v = 0; v < 8; v++) {
        int r = mbase + v;
        if (r < M) {
          float x = acc[ms][ns][v] + bvl;
          if (act == 1)      x = 1.f / (1.f + __expf(-x));
          else if (act == 2) x = fmaxf(x, 0.f);
          int bb = r / Rin, rr = r - bb * Rin;
          out[((size_t)bb * outStride + rowOff + rr) * (size_t)ldOut + n] = x;
        }
      }
    }
  }
}

// ---------------------------------------------------------------------------
// el/er: per-node dot of h row (G=256) with al / ar. One wave per node.
// h has batch row stride HPAD; el/er are [B, N_NODE] contiguous.
// ---------------------------------------------------------------------------
__global__ __launch_bounds__(256) void elr_kernel(
    const float* __restrict__ h, const float* __restrict__ al,
    const float* __restrict__ ar, float* __restrict__ el, float* __restrict__ er)
{
  int gw   = (blockIdx.x * blockDim.x + threadIdx.x) >> 5;
  int lane = threadIdx.x & 31;
  if (gw >= B_ * N_NODE) return;
  int b = gw / N_NODE, r = gw - b * N_NODE;
  const float* hr = h + ((size_t)b * HPAD + r) * G_;
  float sl = 0.f, sr = 0.f;
  #pragma unroll
  for (int i = 0; i < 2; i++) {                 // 2 x (32 lanes x float4) = 256
    int o = (lane + i * 32) * 4;
    v4f v = *(const v4f*)&hr[o];
    v4f a = *(const v4f*)&al[o];
    v4f c = *(const v4f*)&ar[o];
    sl += v[0]*a[0] + v[1]*a[1] + v[2]*a[2] + v[3]*a[3];
    sr += v[0]*c[0] + v[1]*c[1] + v[2]*c[2] + v[3]*c[3];
  }
  #pragma unroll
  for (int off = 16; off > 0; off >>= 1) {
    sl += __shfl_xor(sl, off, 32);
    sr += __shfl_xor(sr, off, 32);
  }
  if (lane == 0) { el[gw] = sl; er[gw] = sr; }
}

// ---------------------------------------------------------------------------
// Fused GAT attention + aggregation: per block, 16 dst rows of one batch.
//  1) masked leaky-relu softmax over 637 srcs -> f16 weights in LDS (pad=0)
//  2) WMMA loop over K=640: out[16 x 256] = a @ h, fused ELU, store.
// 8 waves: wave w owns g columns [w*32, w*32+32). h tiles prefetched;
// h staged as k-strips so transposed LDS stores are packed b64.
// ---------------------------------------------------------------------------
__global__ __launch_bounds__(256) void gat_attn_kernel(
    const float* __restrict__ h, const float* __restrict__ el,
    const float* __restrict__ er, float* __restrict__ gout)
{
  __shared__ float    elS[HPAD];
  __shared__ float    erS[16];
  __shared__ _Float16 aS[16 * ASTR];
  __shared__ _Float16 htS[256 * LSTR];   // h tile transposed [g][k]
  __shared__ float    red[256];
  __shared__ float    rmax[16];
  __shared__ float    rsum[16];

  const int tid = threadIdx.x;
  const int b   = blockIdx.x / 40;
  const int t0  = (blockIdx.x % 40) * 16;

  for (int s = tid; s < HPAD; s += 256)
    elS[s] = (s < N_NODE) ? el[b * N_NODE + s] : -1e30f;
  if (tid < 16) {
    int d = t0 + tid;
    erS[tid] = (d < N_NODE) ? er[b * N_NODE + d] : 0.f;
  }
  __syncthreads();

  const int r  = tid >> 4;        // dst row within tile 0..15
  const int c  = tid & 15;
  const int dg = t0 + r;          // global dst index

  auto score = [&](int s) -> float {
    float v = elS[s] + erS[r];
    float e = (v > 0.f) ? v : NEG_SLOPE * v;
    return (s >= N_NODE || s == dg) ? -1e30f : e;
  };

  // pass 1: row max
  float m = -1e30f;
  for (int s = c; s < HPAD; s += 16) m = fmaxf(m, score(s));
  red[r * 16 + c] = m;
  __syncthreads();
  if (c == 0) {
    float mm = -1e30f;
    for (int i = 0; i < 16; i++) mm = fmaxf(mm, red[r * 16 + i]);
    rmax[r] = mm;
  }
  __syncthreads();
  float rm = rmax[r];

  // pass 2: row sum of exp
  float sum = 0.f;
  for (int s = c; s < HPAD; s += 16) sum += __expf(score(s) - rm);
  red[r * 16 + c] = sum;
  __syncthreads();
  if (c == 0) {
    float ss = 0.f;
    for (int i = 0; i < 16; i++) ss += red[r * 16 + i];
    rsum[r] = ss;
  }
  __syncthreads();
  float inv = 1.f / rsum[r];

  // pass 3: normalized f16 weights into LDS (masked/padded entries -> 0)
  for (int s = c; s < HPAD; s += 16)
    aS[r * ASTR + s] = (_Float16)(__expf(score(s) - rm) * inv);

  const int lane = tid & 31;
  const int wid  = tid >> 5;
  const int koff = (lane & 16) ? 8 : 0;
  const int l15  = lane & 15;
  const int g0   = wid * 32;
  const v8f vzero = {0.f,0.f,0.f,0.f,0.f,0.f,0.f,0.f};
  v8f acc0 = vzero, acc1 = vzero;

  float hRk[8][4];                // 8 k-strips of 4, g fast => coalesced loads
  auto load_ht = [&](int k0) {
    #pragma unroll
    for (int it = 0; it < 8; it++) {
      int i  = tid + it * 256;
      int g  = i & 255, kq = i >> 8;
      #pragma unroll
      for (int j = 0; j < 4; j++)
        hRk[it][j] = h[((size_t)b * HPAD + k0 + kq * 4 + j) * G_ + g];
    }
  };
  auto store_ht = [&]() {
    #pragma unroll
    for (int it = 0; it < 8; it++) {
      int i  = tid + it * 256;
      int g  = i & 255, kq = i >> 8;
      v4f t  = { hRk[it][0], hRk[it][1], hRk[it][2], hRk[it][3] };
      *(v4h*)&htS[g * LSTR + kq * 4] = pack4(t);   // b64 store
    }
  };

  load_ht(0);
  for (int k0 = 0; k0 < HPAD; k0 += 32) {
    __syncthreads();   // covers aS publish (1st iter) + previous htS consumption
    store_ht();
    __syncthreads();
    if (k0 + 32 < HPAD) load_ht(k0 + 32);     // prefetch under WMMAs

    v8h alo = *(const v8h*)&aS[l15 * ASTR + k0 + koff];
    v8h ahi = *(const v8h*)&aS[l15 * ASTR + k0 + 16 + koff];
    v16h af = make_v16(alo, ahi);

    int gc0 = g0 + l15;
    v8h b0l = *(const v8h*)&htS[gc0 * LSTR + koff];
    v8h b0h = *(const v8h*)&htS[gc0 * LSTR + 16 + koff];
    v16h bf0 = make_v16(b0l, b0h);
    int gc1 = g0 + 16 + l15;
    v8h b1l = *(const v8h*)&htS[gc1 * LSTR + koff];
    v8h b1h = *(const v8h*)&htS[gc1 * LSTR + 16 + koff];
    v16h bf1 = make_v16(b1l, b1h);

    acc0 = __builtin_amdgcn_wmma_f32_16x16x32_f16(false, af, false, bf0, (short)0, acc0, false, false);
    acc1 = __builtin_amdgcn_wmma_f32_16x16x32_f16(false, af, false, bf1, (short)0, acc1, false, false);
  }

  // epilogue: ELU, store [16 x 32] per wave
  const int mb = (lane & 16) ? 8 : 0;
  #pragma unroll
  for (int v = 0; v < 8; v++) {
    int d = t0 + mb + v;
    if (d < N_NODE) {
      size_t base = ((size_t)b * N_NODE + d) * G_;
      float x0 = acc0[v]; x0 = (x0 > 0.f) ? x0 : (__expf(x0) - 1.f);
      gout[base + g0 + l15] = x0;
      float x1 = acc1[v]; x1 = (x1 > 0.f) ? x1 : (__expf(x1) - 1.f);
      gout[base + g0 + 16 + l15] = x1;
    }
  }
}

// ---------------------------------------------------------------------------
// Per-batch modality means of shared-branch projections (x = [B,637,384]).
// ---------------------------------------------------------------------------
__global__ __launch_bounds__(384) void mean_kernel(
    const float* __restrict__ x, float* __restrict__ hgi, float* __restrict__ hgt)
{
  int b = blockIdx.x, p = threadIdx.x;
  const float* xb = x + (size_t)b * N_NODE * P_;
  float s = 0.f;
  for (int r = 0; r < I_; r++) s += xb[(size_t)r * P_ + p];
  hgi[b * P_ + p] = s * (1.f / (float)I_);
  s = 0.f;
  for (int r = I_; r < N_NODE; r++) s += xb[(size_t)r * P_ + p];
  hgt[b * P_ + p] = s * (1.f / (float)T_);
}

// ---------------------------------------------------------------------------
// Symmetric KL partials: one block per batch row, writes partial[b].
// ---------------------------------------------------------------------------
__global__ __launch_bounds__(128) void symkl_kernel(
    const float* __restrict__ P, const float* __restrict__ Q,
    int dim, int rstride, float* __restrict__ partial)
{
  __shared__ float red[128];
  int b = blockIdx.x, tid = threadIdx.x;
  const float* p = P + (size_t)b * rstride;
  const float* q = Q + (size_t)b * rstride;

  float mp = -1e30f, mq = -1e30f;
  for (int i = tid; i < dim; i += 128) { mp = fmaxf(mp, p[i]); mq = fmaxf(mq, q[i]); }
  red[tid] = mp; __syncthreads();
  for (int s = 64; s > 0; s >>= 1) { if (tid < s) red[tid] = fmaxf(red[tid], red[tid + s]); __syncthreads(); }
  mp = red[0]; __syncthreads();
  red[tid] = mq; __syncthreads();
  for (int s = 64; s > 0; s >>= 1) { if (tid < s) red[tid] = fmaxf(red[tid], red[tid + s]); __syncthreads(); }
  mq = red[0]; __syncthreads();

  float sp = 0.f, sq = 0.f;
  for (int i = tid; i < dim; i += 128) { sp += __expf(p[i] - mp); sq += __expf(q[i] - mq); }
  red[tid] = sp; __syncthreads();
  for (int s = 64; s > 0; s >>= 1) { if (tid < s) red[tid] += red[tid + s]; __syncthreads(); }
  sp = red[0]; __syncthreads();
  red[tid] = sq; __syncthreads();
  for (int s = 64; s > 0; s >>= 1) { if (tid < s) red[tid] += red[tid + s]; __syncthreads(); }
  sq = red[0]; __syncthreads();

  float lsp = logf(sp), lsq = logf(sq);
  float acc = 0.f;
  for (int i = tid; i < dim; i += 128) {
    float lp = p[i] - mp - lsp;
    float lq = q[i] - mq - lsq;
    acc += 0.5f * (__expf(lq) * (lq - lp) + __expf(lp) * (lp - lq));
  }
  red[tid] = acc; __syncthreads();
  for (int s = 64; s > 0; s >>= 1) { if (tid < s) red[tid] += red[tid + s]; __syncthreads(); }
  if (tid == 0) partial[b] = red[0];
}

// Deterministic fixed-order final sums -> out[0]=share_loss, out[1]=graph_loss
__global__ void finalsum_kernel(const float* __restrict__ part, float* __restrict__ out2)
{
  int t = threadIdx.x;
  if (t < 2) {
    float s = 0.f;
    for (int i = 0; i < B_; i++) s += part[t * 32 + i];
    out2[t] = s;
  }
}

// ---------------------------------------------------------------------------
extern "C" void kernel_launch(void* const* d_in, const int* in_sizes, int n_in,
                              void* d_out, int out_size, void* d_ws, size_t ws_size,
                              hipStream_t stream)
{
  (void)in_sizes; (void)n_in; (void)out_size; (void)ws_size;
  const float* utt_t = (const float*)d_in[0];
  const float* utt_v = (const float*)d_in[1];
  const float* Wpt   = (const float*)d_in[2];
  const float* bpt   = (const float*)d_in[3];
  const float* Wpv   = (const float*)d_in[4];
  const float* bpv   = (const float*)d_in[5];
  const float* Ws    = (const float*)d_in[6];
  const float* bs    = (const float*)d_in[7];
  const float* Wproj = (const float*)d_in[8];
  const float* bproj = (const float*)d_in[9];
  const float* Wg    = (const float*)d_in[10];
  const float* al    = (const float*)d_in[11];
  const float* ar    = (const float*)d_in[12];
  const float* Wc    = (const float*)d_in[13];
  const float* bc    = (const float*)d_in[14];
  float* out = (float*)d_out;

  // scratch layout (floats)
  float* sv  = (float*)d_ws;                           // [B*I, H]
  float* st  = sv  + (size_t)B_ * I_ * H_;             // [B*T, H]
  float* xb  = st  + (size_t)B_ * T_ * H_;             // [B, 637, P]
  float* hb  = xb  + (size_t)B_ * N_NODE * P_;         // [B, 640, G]  (padded)
  float* elb = hb  + (size_t)B_ * HPAD * G_;           // [B, 637]
  float* erb = elb + (size_t)B_ * N_NODE;              // [B, 637]
  float* gb  = erb + (size_t)B_ * N_NODE;              // [B, 637, G]
  float* hgi = gb  + (size_t)B_ * N_NODE * G_;         // [B, P]
  float* hgt = hgi + (size_t)B_ * P_;                  // [B, P]
  float* prt = hgt + (size_t)B_ * P_;                  // [64]

  const size_t E = (size_t)B_ * N_NODE * C_;
  const int MT = B_ * T_;        // 1920
  const int MV = B_ * I_;        // 18464
  const int MN = B_ * N_NODE;    // 20384

  auto gemm = [&](const float* A, const float* W, const float* bias, float* o,
                  int M, int N, int K, int act, int Rin, int oStride, int rOff, int ldo) {
    dim3 grid((M + 127) / 128, N / 128);
    gemm_act_kernel<<<grid, dim3(256), 0, stream>>>(A, W, bias, o, M, N, K, act,
                                                    Rin, oStride, rOff, ldo);
  };

  for (int br = 0; br < 2; br++) {
    const float* Wt = br ? Ws : Wpt; const float* bt = br ? bs : bpt;
    const float* Wv = br ? Ws : Wpv; const float* bv = br ? bs : bpv;
    float* emb = out + (br ? E : 0);

    // sigmoid projections
    gemm(utt_t, Wt, bt, st, MT, H_, H_, /*sigmoid*/1, MT, MT, 0, H_);
    gemm(utt_v, Wv, bv, sv, MV, H_, H_, /*sigmoid*/1, MV, MV, 0, H_);
    // relu proj, written directly into concatenated x [B,637,P]
    gemm(sv, Wproj, bproj, xb, MV, P_, H_, /*relu*/2, I_, N_NODE, 0,  P_);
    gemm(st, Wproj, bproj, xb, MT, P_, H_, /*relu*/2, T_, N_NODE, I_, P_);
    // h = x @ Wg  (no bias/act), padded batch stride 640 rows
    gemm(xb, Wg, nullptr, hb, MN, G_, P_, 0, N_NODE, HPAD, 0, G_);
    // attention scores + softmax + aggregation + ELU
    elr_kernel<<<(MN * 32 + 255) / 256, 256, 0, stream>>>(hb, al, ar, elb, erb);
    gat_attn_kernel<<<B_ * 40, 256, 0, stream>>>(hb, elb, erb, gb);
    // emb = g @ Wc + bc  -> output region
    gemm(gb, Wc, bc, emb, MN, C_, G_, 0, N_NODE, N_NODE, 0, C_);

    if (br == 1) mean_kernel<<<B_, 384, 0, stream>>>(xb, hgi, hgt);
  }

  // share_loss: symKL(hg_i, hg_t); graph_loss: symKL(share CLS, private CLS)
  symkl_kernel<<<B_, 128, 0, stream>>>(hgi, hgt, P_, P_, prt);
  symkl_kernel<<<B_, 128, 0, stream>>>(out + E, out, C_, N_NODE * C_, prt + 32);
  finalsum_kernel<<<1, 64, 0, stream>>>(prt, out + 2 * E);
}
